// TransformerBlock_58497454572197
// MI455X (gfx1250) — matrix operand order repaired and checked by reading, not compiled, not run on hardware
//
#include <hip/hip_runtime.h>

#define D_MODEL 768
#define N_HEADS 12
#define HEAD_DIM 64
#define D_FF    3072
#define SEQ     2048
#define BATCH   2
#define ROWS    (BATCH*SEQ)   // 4096

typedef _Float16 h16;
typedef __attribute__((ext_vector_type(8)))  _Float16 v8h;
typedef __attribute__((ext_vector_type(16))) _Float16 v16h;
typedef __attribute__((ext_vector_type(8)))  float    v8f;
typedef __attribute__((ext_vector_type(4)))  unsigned int u32x4;
typedef __attribute__((ext_vector_type(8)))  int i32x8;
typedef __attribute__((ext_vector_type(4)))  int i32x4;

#if __has_builtin(__builtin_amdgcn_tensor_load_to_lds)
#define HAVE_TDM 1
#else
#define HAVE_TDM 0
#endif

#if __has_builtin(__builtin_amdgcn_s_wait_tensorcnt)
#define WAIT_TENSOR() __builtin_amdgcn_s_wait_tensorcnt(0)
#else
#define WAIT_TENSOR() asm volatile("s_wait_tensorcnt 0" ::: "memory")
#endif

// ---------------------------------------------------------------- WMMA utils
static __device__ __forceinline__ v8f wmma16(v16h a, v16h b, v8f c) {
  // D = A(16x32 f16) * B(32x16 f16) + C(16x16 f32)
  return __builtin_amdgcn_wmma_f32_16x16x32_f16(false, a, false, b, (short)0, c,
                                                false, false);
}

// A-fragment (16x32 f16). Lane m = lane%16 holds row m.
// Layout: element e -> k = e + half*16 (contiguous, matches B-fragment).
static __device__ __forceinline__ v16h load_a_frag(const h16* rowk, int hl) {
  v8h lo = *(const v8h*)(rowk + hl * 16);
  v8h hi = *(const v8h*)(rowk + hl * 16 + 8);
  return __builtin_shufflevector(lo, hi, 0, 1, 2, 3, 4, 5, 6, 7,
                                 8, 9, 10, 11, 12, 13, 14, 15);
}

// B-fragment (32x16 f16) from K-major storage Bt[n][k]: lanes 0-15 hold
// K=0..15, lanes 16-31 hold K=16..31 -> 16 contiguous k at half*16 (32B load).
static __device__ __forceinline__ v16h load_b_frag(const h16* rowk, int hl) {
  return *(const v16h*)(rowk + hl * 16);
}

// ------------------------------------------------------- TDM 2D tile -> LDS
// Loads a tile of nrows x 32 f16 (rows stride K elements) into LDS packed as
// [nrows][32]. D# packed per CDNA5 ISA 8.3/8.4 (count=1, type=2, dsz=2B).
static __device__ __forceinline__ void tdm_tile_load(unsigned lds_off,
                                                     const h16* gtile, int K,
                                                     int nrows) {
#if HAVE_TDM
  unsigned long long ga = (unsigned long long)(uintptr_t)gtile;
  u32x4 g0;
  g0[0] = 1u;                                      // count=1, user mode
  g0[1] = lds_off;                                 // lds_addr (bytes)
  g0[2] = (unsigned)ga;                            // global_addr[31:0]
  g0[3] = (unsigned)((ga >> 32) & 0x01ffffffull) | 0x80000000u;  // type=2
  i32x8 g1;
  g1[0] = 0x00010000;                              // data_size = 2 bytes
  g1[1] = (int)(((unsigned)K & 0xffffu) << 16);    // tensor_dim0[15:0]
  g1[2] = (int)((((unsigned)K >> 16) & 0xffffu) |  // tensor_dim0[31:16]
                (((unsigned)nrows & 0xffffu) << 16));  // tensor_dim1[15:0]
  g1[3] = (int)((32u << 16));                      // tile_dim0 = 32
  g1[4] = nrows & 0xffff;                          // tile_dim1, tile_dim2=0
  g1[5] = K;                                       // tensor_dim0_stride lo
  g1[6] = 0;
  g1[7] = 0;
  i32x4 gz4 = {0, 0, 0, 0};                        // D# groups 2/3: unused (2D)
  i32x8 gz8 = {0, 0, 0, 0, 0, 0, 0, 0};
  __builtin_amdgcn_tensor_load_to_lds(g0, g1, gz4, gz4, gz8, 0);
#else
  (void)lds_off; (void)gtile; (void)K; (void)nrows;
#endif
}

// Fallback cooperative copy: 128 rows x 32 halves, 256 threads x 16 halves.
static __device__ __forceinline__ void coop_copy_tile(h16* dst,
                                                      const h16* gsrc, int K,
                                                      int tid) {
  const int r = tid >> 1;
  const int c = (tid & 1) * 16;
  const v8h* s = (const v8h*)(gsrc + (size_t)r * K + c);
  v8h* d = (v8h*)(dst + r * 32 + c);
  d[0] = s[0];
  d[1] = s[1];
}

// ---------------------------------------------------------------- weight prep
__global__ __launch_bounds__(256) void transpose_cast_f16(
    const float* __restrict__ W, h16* __restrict__ Wt, int K, int N) {
  int id = blockIdx.x * 256 + threadIdx.x;
  if (id < K * N) {
    int k = id / N, n = id - k * N;
    Wt[(size_t)n * K + k] = (h16)W[id];
  }
}

// ---------------------------------------------------------------- layernorm
__global__ __launch_bounds__(256) void ln_f16(
    const float* __restrict__ X, const float* __restrict__ G,
    const float* __restrict__ Bb, h16* __restrict__ Y) {
  __shared__ float red[256];
  const int t = threadIdx.x;
  const size_t row = blockIdx.x;
  const float* x = X + row * D_MODEL;
  float v0 = x[t], v1 = x[t + 256], v2 = x[t + 512];
  red[t] = v0 + v1 + v2;
  __syncthreads();
  for (int o = 128; o > 0; o >>= 1) {
    if (t < o) red[t] += red[t + o];
    __syncthreads();
  }
  float mu = red[0] * (1.0f / D_MODEL);
  __syncthreads();
  float d0 = v0 - mu, d1 = v1 - mu, d2 = v2 - mu;
  red[t] = d0 * d0 + d1 * d1 + d2 * d2;
  __syncthreads();
  for (int o = 128; o > 0; o >>= 1) {
    if (t < o) red[t] += red[t + o];
    __syncthreads();
  }
  float rstd = rsqrtf(red[0] * (1.0f / D_MODEL) + 1e-5f);
  h16* y = Y + row * D_MODEL;
  y[t]       = (h16)(d0 * rstd * G[t]       + Bb[t]);
  y[t + 256] = (h16)(d1 * rstd * G[t + 256] + Bb[t + 256]);
  y[t + 512] = (h16)(d2 * rstd * G[t + 512] + Bb[t + 512]);
}

// ---------------------------------------------------------------- GEMM
// C[M,N] = A[M,K](f16,row-major) * Bt[N,K](f16,K-major) + bias, with epilogues.
// Block 256 thr = 8 waves; block tile 128x128; wave tile 32x64.
// A/B K-slabs (128x32 f16) DMA'd to LDS by the Tensor Data Mover, dbl-buffered.
#define MODE_QK     0   // f16 out, [B][H][S][64]
#define MODE_V      1   // f16 out, [B][H][64][S]  (transposed for P*V)
#define MODE_F32RES 2   // f32 out, + residual
#define MODE_RELU16 3   // f16 out, relu

__global__ __launch_bounds__(256) void gemm_wmma(
    const h16* __restrict__ A, const h16* __restrict__ Bt,
    const float* __restrict__ bias, const float* __restrict__ res,
    void* __restrict__ out, int M, int N, int K, int mode) {
  __shared__ __align__(128) h16 Asb[2][128][32];
  __shared__ __align__(128) h16 Bsb[2][128][32];

  const int tid  = threadIdx.x;
  const int lane = tid & 31;
  const int w    = tid >> 5;
  const int hl   = lane >> 4;
  const int l16  = lane & 15;
  const int rowBase = blockIdx.x * 128 + (w & 3) * 32;
  const int colBase = blockIdx.y * 128 + (w >> 2) * 64;

  const h16* Ablk = A + (size_t)(blockIdx.x * 128) * K;
  const h16* Bblk = Bt + (size_t)(blockIdx.y * 128) * K;

#if HAVE_TDM
  const bool issuer = (tid < 32);
  if (issuer) {
    tdm_tile_load((unsigned)(uintptr_t)&Asb[0][0][0], Ablk, K, 128);
    tdm_tile_load((unsigned)(uintptr_t)&Bsb[0][0][0], Bblk, K, 128);
    WAIT_TENSOR();
  }
#else
  coop_copy_tile(&Asb[0][0][0], Ablk, K, tid);
  coop_copy_tile(&Bsb[0][0][0], Bblk, K, tid);
#endif
  __syncthreads();

  v8f acc[8] = {};
  const int aLocal = (w & 3) * 32 + l16;
  const int bLocal = (w >> 2) * 64 + l16;

  int cur = 0;
  for (int kb = 0; kb < K; kb += 32, cur ^= 1) {
    const int nxt = cur ^ 1;
    // Kick off DMA of the next K-slab while this one is consumed.
    if (kb + 32 < K) {
#if HAVE_TDM
      if (issuer) {
        tdm_tile_load((unsigned)(uintptr_t)&Asb[nxt][0][0],
                      Ablk + (kb + 32), K, 128);
        tdm_tile_load((unsigned)(uintptr_t)&Bsb[nxt][0][0],
                      Bblk + (kb + 32), K, 128);
      }
#else
      coop_copy_tile(&Asb[nxt][0][0], Ablk + (kb + 32), K, tid);
      coop_copy_tile(&Bsb[nxt][0][0], Bblk + (kb + 32), K, tid);
#endif
    }

    const v16h a0 = load_a_frag(&Asb[cur][aLocal][0], hl);
    const v16h a1 = load_a_frag(&Asb[cur][aLocal + 16][0], hl);
#pragma unroll
    for (int g = 0; g < 4; ++g) {
      const v16h bf = load_b_frag(&Bsb[cur][bLocal + g * 16][0], hl);
      acc[g]     = wmma16(a0, bf, acc[g]);
      acc[g + 4] = wmma16(a1, bf, acc[g + 4]);
    }

#if HAVE_TDM
    if (issuer) WAIT_TENSOR();   // next slab landed before anyone reads it
#endif
    __syncthreads();             // also protects WAR on buffer reuse
  }

  float bv[4];
  int col[4];
#pragma unroll
  for (int g = 0; g < 4; ++g) {
    col[g] = colBase + g * 16 + l16;
    bv[g] = bias[col[g]];
  }

#pragma unroll
  for (int mh = 0; mh < 2; ++mh) {
#pragma unroll
    for (int r = 0; r < 8; ++r) {
      const int row = rowBase + mh * 16 + r + 8 * hl;  // C layout
#pragma unroll
      for (int g = 0; g < 4; ++g) {
        float v = acc[mh * 4 + g][r] + bv[g];
        if (mode == MODE_QK) {
          int b_ = row >> 11, tt = row & (SEQ - 1);
          int hh = col[g] >> 6, dd = col[g] & (HEAD_DIM - 1);
          ((h16*)out)[(((size_t)(b_ * N_HEADS + hh) << 11) + tt) * HEAD_DIM +
                      dd] = (h16)v;
        } else if (mode == MODE_V) {
          int b_ = row >> 11, tt = row & (SEQ - 1);
          int hh = col[g] >> 6, dd = col[g] & (HEAD_DIM - 1);
          ((h16*)out)[((size_t)(b_ * N_HEADS + hh) * HEAD_DIM + dd) * SEQ +
                      tt] = (h16)v;
        } else if (mode == MODE_RELU16) {
          ((h16*)out)[(size_t)row * N + col[g]] = (h16)fmaxf(v, 0.0f);
        } else {  // MODE_F32RES
          ((float*)out)[(size_t)row * N + col[g]] =
              v + res[(size_t)row * N + col[g]];
        }
      }
    }
  }
}

// ---------------------------------------------------------------- attention
// One wave per 16-query tile per (batch, head). Flash-style online softmax.
// Q,K in [B][H][S][64] f16 ; Vt in [B][H][64][S] f16 ; out f16 [B*S][768].
__global__ __launch_bounds__(128) void attn_wmma(
    const h16* __restrict__ Q, const h16* __restrict__ Kq,
    const h16* __restrict__ Vt, h16* __restrict__ O) {
  __shared__ __align__(32) h16 Pl[4][16][40];  // per-wave P staging

  const int lane = threadIdx.x & 31;
  const int wib  = threadIdx.x >> 5;
  const int hl   = lane >> 4;
  const int l16  = lane & 15;

  const int wid = blockIdx.x * 4 + wib;
  const int qt = wid & 127;
  const int h  = (wid >> 7) % N_HEADS;
  const int b  = wid / (128 * N_HEADS);
  const size_t bhS = (size_t)(b * N_HEADS + h) * SEQ;

  const h16* qRow = Q + (bhS + (size_t)qt * 16 + l16) * HEAD_DIM;
  const v16h qf0 = load_a_frag(qRow, hl);
  const v16h qf1 = load_a_frag(qRow + 32, hl);

  v8f o[4] = {};
  float rm[8], rs[8];
#pragma unroll
  for (int r = 0; r < 8; ++r) { rm[r] = -1e30f; rs[r] = 0.0f; }

  const int kbMax = (qt * 16 + 15) >> 5;
  const h16* vBase = Vt + ((size_t)(b * N_HEADS + h) * HEAD_DIM + l16) * SEQ;

  for (int kb = 0; kb <= kbMax; ++kb) {
    v8f s0 = {}, s1 = {};
    const h16* kRow0 = Kq + (bhS + (size_t)kb * 32 + l16) * HEAD_DIM;
    const h16* kRow1 = kRow0 + (size_t)16 * HEAD_DIM;
    s0 = wmma16(qf0, load_b_frag(kRow0, hl), s0);
    s0 = wmma16(qf1, load_b_frag(kRow0 + 32, hl), s0);
    s1 = wmma16(qf0, load_b_frag(kRow1, hl), s1);
    s1 = wmma16(qf1, load_b_frag(kRow1 + 32, hl), s1);

    const bool maskblk = (kb == kbMax);
#pragma unroll
    for (int r = 0; r < 8; ++r) {
      float x0 = s0[r] * 0.125f;   // 1/sqrt(64)
      float x1 = s1[r] * 0.125f;
      if (maskblk) {
        const int qi = qt * 16 + r + 8 * hl;
        const int k0 = kb * 32 + l16;
        if (k0 > qi) x0 = -1e30f;
        if (k0 + 16 > qi) x1 = -1e30f;
      }
      float tmax = fmaxf(x0, x1);
#pragma unroll
      for (int off = 1; off < 16; off <<= 1)
        tmax = fmaxf(tmax, __shfl_xor(tmax, off, 32));
      const float nm = fmaxf(rm[r], tmax);
      const float resc = __expf(rm[r] - nm);
      const float p0 = __expf(x0 - nm);
      const float p1 = __expf(x1 - nm);
      float psum = p0 + p1;
#pragma unroll
      for (int off = 1; off < 16; off <<= 1)
        psum += __shfl_xor(psum, off, 32);
      rs[r] = rs[r] * resc + psum;
      rm[r] = nm;
      o[0][r] *= resc; o[1][r] *= resc; o[2][r] *= resc; o[3][r] *= resc;
      Pl[wib][r + 8 * hl][l16]      = (h16)p0;   // C layout -> A layout
      Pl[wib][r + 8 * hl][l16 + 16] = (h16)p1;
    }
    asm volatile("s_wait_dscnt 0" ::: "memory");

    const v16h pf = load_a_frag(&Pl[wib][l16][0], hl);
    const h16* vRow = vBase + (size_t)kb * 32;
#pragma unroll
    for (int g = 0; g < 4; ++g)
      o[g] = wmma16(pf, load_b_frag(vRow + (size_t)g * 16 * SEQ, hl), o[g]);
    asm volatile("s_wait_dscnt 0" ::: "memory");
  }

#pragma unroll
  for (int r = 0; r < 8; ++r) {
    const float inv = 1.0f / rs[r];
    const int t = qt * 16 + r + 8 * hl;
    const size_t base = ((size_t)b * SEQ + t) * D_MODEL + h * HEAD_DIM;
#pragma unroll
    for (int g = 0; g < 4; ++g)
      O[base + g * 16 + l16] = (h16)(o[g][r] * inv);
  }
}

// ---------------------------------------------------------------- launcher
extern "C" void kernel_launch(void* const* d_in, const int* in_sizes, int n_in,
                              void* d_out, int out_size, void* d_ws,
                              size_t ws_size, hipStream_t stream) {
  (void)in_sizes; (void)n_in; (void)out_size; (void)ws_size;
  const float* x    = (const float*)d_in[0];
  const float* wq   = (const float*)d_in[1];
  const float* bq   = (const float*)d_in[2];
  const float* wk   = (const float*)d_in[3];
  const float* bk   = (const float*)d_in[4];
  const float* wv   = (const float*)d_in[5];
  const float* bv   = (const float*)d_in[6];
  const float* wo   = (const float*)d_in[7];
  const float* bo   = (const float*)d_in[8];
  const float* w1   = (const float*)d_in[9];
  const float* b1   = (const float*)d_in[10];
  const float* w2   = (const float*)d_in[11];
  const float* b2   = (const float*)d_in[12];
  const float* ln1g = (const float*)d_in[13];
  const float* ln1b = (const float*)d_in[14];
  const float* ln2g = (const float*)d_in[15];
  const float* ln2b = (const float*)d_in[16];

  char* p = (char*)d_ws;
  auto carve = [&](size_t bytes) -> void* {
    void* r = (void*)p;
    p += (bytes + 255) & ~(size_t)255;
    return r;
  };
  const size_t DD = (size_t)D_MODEL * D_MODEL;
  const size_t DF = (size_t)D_MODEL * D_FF;
  const size_t RD = (size_t)ROWS * D_MODEL;
  const size_t RF = (size_t)ROWS * D_FF;

  h16* wqT = (h16*)carve(DD * 2);
  h16* wkT = (h16*)carve(DD * 2);
  h16* wvT = (h16*)carve(DD * 2);
  h16* woT = (h16*)carve(DD * 2);
  h16* w1T = (h16*)carve(DF * 2);
  h16* w2T = (h16*)carve(DF * 2);
  h16* xln = (h16*)carve(RD * 2);
  h16* qb  = (h16*)carve(RD * 2);
  h16* kb  = (h16*)carve(RD * 2);
  h16* vtb = (h16*)carve(RD * 2);
  h16* ab  = (h16*)carve(RD * 2);
  h16* h2  = (h16*)carve(RD * 2);
  h16* ffb = (h16*)carve(RF * 2);
  float* x1 = (float*)carve(RD * 4);

  const int gDD = (int)((DD + 255) / 256);
  const int gDF = (int)((DF + 255) / 256);
  transpose_cast_f16<<<gDD, 256, 0, stream>>>(wq, wqT, D_MODEL, D_MODEL);
  transpose_cast_f16<<<gDD, 256, 0, stream>>>(wk, wkT, D_MODEL, D_MODEL);
  transpose_cast_f16<<<gDD, 256, 0, stream>>>(wv, wvT, D_MODEL, D_MODEL);
  transpose_cast_f16<<<gDD, 256, 0, stream>>>(wo, woT, D_MODEL, D_MODEL);
  transpose_cast_f16<<<gDF, 256, 0, stream>>>(w1, w1T, D_MODEL, D_FF);
  transpose_cast_f16<<<gDF, 256, 0, stream>>>(w2, w2T, D_FF, D_MODEL);

  ln_f16<<<ROWS, 256, 0, stream>>>(x, ln1g, ln1b, xln);

  dim3 g768(ROWS / 128, D_MODEL / 128);
  dim3 g3072(ROWS / 128, D_FF / 128);

  gemm_wmma<<<g768, 256, 0, stream>>>(xln, wqT, bq, nullptr, qb,
                                      ROWS, D_MODEL, D_MODEL, MODE_QK);
  gemm_wmma<<<g768, 256, 0, stream>>>(xln, wkT, bk, nullptr, kb,
                                      ROWS, D_MODEL, D_MODEL, MODE_QK);
  gemm_wmma<<<g768, 256, 0, stream>>>(xln, wvT, bv, nullptr, vtb,
                                      ROWS, D_MODEL, D_MODEL, MODE_V);

  attn_wmma<<<(BATCH * N_HEADS * (SEQ / 16)) / 4, 128, 0, stream>>>(qb, kb,
                                                                    vtb, ab);

  gemm_wmma<<<g768, 256, 0, stream>>>(ab, woT, bo, x, x1,
                                      ROWS, D_MODEL, D_MODEL, MODE_F32RES);

  ln_f16<<<ROWS, 256, 0, stream>>>(x1, ln2g, ln2b, h2);

  gemm_wmma<<<g3072, 256, 0, stream>>>(h2, w1T, b1, nullptr, ffb,
                                       ROWS, D_FF, D_MODEL, MODE_RELU16);
  gemm_wmma<<<g768, 256, 0, stream>>>(ffb, w2T, b2, x1, d_out,
                                      ROWS, D_MODEL, D_FF, MODE_F32RES);
}
